// MultiHeadSelfAttentionBlock_41351945126215
// MI455X (gfx1250) — matrix operand order repaired and checked by reading, not compiled, not run on hardware
//
#include <hip/hip_runtime.h>
#include <hip/hip_bf16.h>

typedef __attribute__((ext_vector_type(16))) _Float16 v16h;
typedef __attribute__((ext_vector_type(8)))  float    v8f;
typedef __attribute__((ext_vector_type(2)))  _Float16 h2;

#define BB 2
#define NN 2048
#define CC 1024
#define HH 16
#define DD 64

// ---------------- attention (flash-style, fused softmax) ----------------
// grid = B*H*(N/64), block = 128 threads = 4 waves; each wave: 16 query rows.
// Each iteration consumes a 64-key block: 18 WMMAs/wave (8 QK + 8 PV + 2 rowsum).
#define QROWS 64
#define KBLK  64
#define KPITCH 68   // Ks row pitch (d) in f16: 64 + 4 pad
#define VPITCH 68   // Vt row pitch (key) in f16: 64 + 4 pad
#define PPITCH 68   // Ps row pitch (key) in f16: 64 + 4 pad

__global__ __launch_bounds__(128) void attn_kernel(
    const float* __restrict__ q, const float* __restrict__ kg,
    const float* __restrict__ vg, _Float16* __restrict__ xout)
{
    __shared__ _Float16 Ks[KBLK * KPITCH];     // [key][d]
    __shared__ _Float16 Vt[DD * VPITCH];       // [d][key]  (transposed for B-frag reads)
    __shared__ _Float16 Ps[4 * 16 * PPITCH];   // per-wave P scratch [row][key]

    const int nqb = NN / QROWS;
    const int bh  = blockIdx.x / nqb;
    const int qb  = blockIdx.x % nqb;
    const int b   = bh / HH, h = bh % HH;
    const int tid = threadIdx.x;
    const int wave = tid >> 5, lane = tid & 31;
    const int half = lane >> 4, col = lane & 15;
    const int qrow0 = qb * QROWS + wave * 16;

    // ---- Q A-fragments: 16x64 = two 16x32 f16 fragments, fp32->f16 on the fly
    // A layout (16-bit, 16x32): lane row = lane%16; VGPR v holds K pair at
    //   kb = (v>=4 ? 16 : 0) + (lane/16)*8 + 2*(v%4)
    v16h aq[2];
    {
        const float* qbase = q + ((size_t)b * NN + (qrow0 + col)) * CC + h * DD;
        #pragma unroll
        for (int g = 0; g < 2; ++g) {
            #pragma unroll
            for (int v = 0; v < 8; ++v) {
                const int kb = ((v >> 2) << 4) + half * 8 + ((v & 3) << 1);
                float2 f = *(const float2*)(qbase + g * 32 + kb);
                aq[g][2 * v]     = (_Float16)f.x;
                aq[g][2 * v + 1] = (_Float16)f.y;
            }
        }
    }

    // all-ones B tile: P x ones = row-sum broadcast to every column/lane
    v16h ones;
    #pragma unroll
    for (int i = 0; i < 16; ++i) ones[i] = (_Float16)1.0f;

    v8f acc[4], sumacc;
    {
        v8f z = {0.f, 0.f, 0.f, 0.f, 0.f, 0.f, 0.f, 0.f};
        #pragma unroll
        for (int dt = 0; dt < 4; ++dt) acc[dt] = z;
        sumacc = z;
    }
    float mrow[8];
    #pragma unroll
    for (int r = 0; r < 8; ++r) mrow[r] = -3.0e38f;

    const float* kbase = kg + ((size_t)(b * HH + h)) * NN * DD;
    const float* vbase = vg + ((size_t)(b * HH + h)) * NN * DD;

    for (int k0 = 0; k0 < NN; k0 += KBLK) {
        __syncthreads();  // protect LDS reuse from previous iteration

        // ---- cooperative fp32->f16 stage: K row-major, V transposed (paired)
        {
            // K: 64 keys x 64 d; thread = half a key row (32 d)
            const int key  = tid >> 1;
            const int dseg = (tid & 1) << 5;
            const float* kr = kbase + (size_t)(k0 + key) * DD + dseg;
            if (k0 + KBLK < NN) {               // global_prefetch_b8 next block
                __builtin_prefetch(kr + (size_t)KBLK * DD, 0, 1);
            }
            float4 kf[8];
            #pragma unroll
            for (int i = 0; i < 8; ++i) kf[i] = ((const float4*)kr)[i];
            #pragma unroll
            for (int i = 0; i < 8; ++i) {
                h2 p0 = {(_Float16)kf[i].x, (_Float16)kf[i].y};
                h2 p1 = {(_Float16)kf[i].z, (_Float16)kf[i].w};
                *(h2*)&Ks[key * KPITCH + dseg + 4 * i]     = p0;
                *(h2*)&Ks[key * KPITCH + dseg + 4 * i + 2] = p1;
            }
        }
        {
            // V: thread loads 16 d-values of two adjacent keys, stores h2 pairs
            const int kp   = tid & 31;          // key pair index -> keys 2kp, 2kp+1
            const int dseg = (tid >> 5) << 4;   // 16 d per wave
            const float* v0 = vbase + (size_t)(k0 + 2 * kp) * DD + dseg;
            const float* v1 = v0 + DD;
            if (k0 + KBLK < NN) {
                __builtin_prefetch(v0 + (size_t)KBLK * DD, 0, 1);
            }
            float4 a4[4], b4[4];
            #pragma unroll
            for (int i = 0; i < 4; ++i) { a4[i] = ((const float4*)v0)[i]; b4[i] = ((const float4*)v1)[i]; }
            float av[16], bv[16];
            #pragma unroll
            for (int i = 0; i < 4; ++i) {
                av[4*i+0]=a4[i].x; av[4*i+1]=a4[i].y; av[4*i+2]=a4[i].z; av[4*i+3]=a4[i].w;
                bv[4*i+0]=b4[i].x; bv[4*i+1]=b4[i].y; bv[4*i+2]=b4[i].z; bv[4*i+3]=b4[i].w;
            }
            #pragma unroll
            for (int i = 0; i < 16; ++i) {
                h2 p = {(_Float16)av[i], (_Float16)bv[i]};
                *(h2*)&Vt[(dseg + i) * VPITCH + 2 * kp] = p;
            }
        }
        __syncthreads();

        // ---- S = Q · K^T : four 16x16 tiles, K-dim 64 => 2 chained WMMAs each
        // B layout (16-bit, 32x16): lane col = lane%16; VGPR v holds K pair at
        //   (v>=4 ? 16 : 0) + (lane/16)*8 + 2*(v%4)
        v8f s[4];
        #pragma unroll
        for (int t = 0; t < 4; ++t) {
            v8f c = {0.f, 0.f, 0.f, 0.f, 0.f, 0.f, 0.f, 0.f};
            #pragma unroll
            for (int g = 0; g < 2; ++g) {
                v16h bf;
                const int key = t * 16 + col;
                #pragma unroll
                for (int v = 0; v < 8; ++v) {
                    const int d = g * 32 + ((v >> 2) << 4) + half * 8 + ((v & 3) << 1);
                    bf[2 * v]     = Ks[key * KPITCH + d];
                    bf[2 * v + 1] = Ks[key * KPITCH + d + 1];
                }
                c = __builtin_amdgcn_wmma_f32_16x16x32_f16(
                        false, aq[g], false, bf, (short)0, c, false, false);
            }
            s[t] = c;
        }

        // ---- online softmax (C layout: lane holds rows r+8*half, col = lane%16)
        _Float16 pf[4][8];
        #pragma unroll
        for (int r = 0; r < 8; ++r) {
            float mx = fmaxf(fmaxf(s[0][r], s[1][r]), fmaxf(s[2][r], s[3][r]));
            #pragma unroll
            for (int off = 1; off < 16; off <<= 1)
                mx = fmaxf(mx, __shfl_xor(mx, off, 32));   // stays inside 16-lane half
            const float mnew  = fmaxf(mrow[r], mx);
            const float scale = __expf(mrow[r] - mnew);
            mrow[r] = mnew;
            sumacc[r] *= scale;
            #pragma unroll
            for (int dt = 0; dt < 4; ++dt) acc[dt][r] *= scale;
            #pragma unroll
            for (int t = 0; t < 4; ++t)
                pf[t][r] = (_Float16)__expf(s[t][r] - mnew);
        }

        // ---- reshape P: C layout -> A layout via per-wave LDS scratch.
        // Wave-private region: no block barrier needed, only a DScnt drain.
        _Float16* myP = &Ps[wave * 16 * PPITCH];
        #pragma unroll
        for (int t = 0; t < 4; ++t)
            #pragma unroll
            for (int r = 0; r < 8; ++r)
                myP[(r + 8 * half) * PPITCH + t * 16 + col] = pf[t][r];
        asm volatile("s_wait_dscnt 0x0" ::: "memory");  // LDS in-order within wave

        v16h pa[2];
        #pragma unroll
        for (int gk = 0; gk < 2; ++gk)
            #pragma unroll
            for (int v = 0; v < 8; ++v) {
                const int kb = gk * 32 + ((v >> 2) << 4) + half * 8 + ((v & 3) << 1);
                pa[gk][2 * v]     = myP[col * PPITCH + kb];
                pa[gk][2 * v + 1] = myP[col * PPITCH + kb + 1];
            }

        // ---- O += P · V  and  rowsum += P · 1  (sum on the matrix pipe)
        #pragma unroll
        for (int gk = 0; gk < 2; ++gk) {
            sumacc = __builtin_amdgcn_wmma_f32_16x16x32_f16(
                         false, pa[gk], false, ones, (short)0, sumacc, false, false);
            #pragma unroll
            for (int dt = 0; dt < 4; ++dt) {
                v16h vb;
                const int d = dt * 16 + col;
                #pragma unroll
                for (int v = 0; v < 8; ++v) {
                    const int key = gk * 32 + ((v >> 2) << 4) + half * 8 + ((v & 3) << 1);
                    vb[2 * v]     = Vt[d * VPITCH + key];
                    vb[2 * v + 1] = Vt[d * VPITCH + key + 1];
                }
                acc[dt] = __builtin_amdgcn_wmma_f32_16x16x32_f16(
                              false, pa[gk], false, vb, (short)0, acc[dt], false, false);
            }
        }
    }

    // ---- normalize and store f16 [B,N,C] (heads re-interleaved into C)
    #pragma unroll
    for (int r = 0; r < 8; ++r) {
        const float inv = 1.0f / sumacc[r];
        const int n = qrow0 + r + 8 * half;
        _Float16* orow = xout + ((size_t)b * NN + n) * CC + h * DD;
        #pragma unroll
        for (int dt = 0; dt < 4; ++dt)
            orow[dt * 16 + col] = (_Float16)(acc[dt][r] * inv);
    }
}

// ---------------- projection GEMM: out[m][co] = x[m][:] · W[co][:] + bias ----
// M = B*N = 4096, N = C = 1024, K = C = 1024. 64x64 tile per block, 4 waves.
#define WPITCH 36

__global__ __launch_bounds__(128) void proj_kernel(
    const _Float16* __restrict__ x, const float* __restrict__ w,
    const float* __restrict__ bias, float* __restrict__ out)
{
    __shared__ _Float16 Ws[64 * WPITCH];  // [co_local][ci_local], fp32->f16 staged

    const int tid = threadIdx.x;
    const int wave = tid >> 5, lane = tid & 31;
    const int half = lane >> 4, col = lane & 15;
    const int row0 = blockIdx.x * 64 + wave * 16;
    const int col0 = blockIdx.y * 64;

    v8f acc[4];
    #pragma unroll
    for (int ct = 0; ct < 4; ++ct) {
        const float bv = bias[col0 + ct * 16 + col];
        v8f t = {bv, bv, bv, bv, bv, bv, bv, bv};
        acc[ct] = t;
    }

    const _Float16* arow = x + (size_t)(row0 + col) * CC;

    for (int k0 = 0; k0 < CC; k0 += 32) {
        __syncthreads();
        {   // stage 64x32 tile of W, fp32 -> f16
            const int co    = tid >> 1;
            const int ciseg = (tid & 1) << 4;
            const float* wr = w + (size_t)(col0 + co) * CC + k0 + ciseg;
            float4 wf[4];
            #pragma unroll
            for (int i = 0; i < 4; ++i) wf[i] = ((const float4*)wr)[i];
            #pragma unroll
            for (int i = 0; i < 4; ++i) {
                h2 p0 = {(_Float16)wf[i].x, (_Float16)wf[i].y};
                h2 p1 = {(_Float16)wf[i].z, (_Float16)wf[i].w};
                *(h2*)&Ws[co * WPITCH + ciseg + 4 * i]     = p0;
                *(h2*)&Ws[co * WPITCH + ciseg + 4 * i + 2] = p1;
            }
        }
        __syncthreads();

        v16h af;
        #pragma unroll
        for (int v = 0; v < 8; ++v) {
            const int kb = ((v >> 2) << 4) + half * 8 + ((v & 3) << 1);
            af[2 * v]     = arow[k0 + kb];
            af[2 * v + 1] = arow[k0 + kb + 1];
        }
        #pragma unroll
        for (int ct = 0; ct < 4; ++ct) {
            v16h bf;
            const int cr = (ct * 16 + col) * WPITCH;
            #pragma unroll
            for (int v = 0; v < 8; ++v) {
                const int ci = ((v >> 2) << 4) + half * 8 + ((v & 3) << 1);
                bf[2 * v]     = Ws[cr + ci];
                bf[2 * v + 1] = Ws[cr + ci + 1];
            }
            acc[ct] = __builtin_amdgcn_wmma_f32_16x16x32_f16(
                          false, af, false, bf, (short)0, acc[ct], false, false);
        }
    }

    #pragma unroll
    for (int r = 0; r < 8; ++r) {
        const int m = row0 + r + 8 * half;
        float* orow = out + (size_t)m * CC + col0;
        #pragma unroll
        for (int ct = 0; ct < 4; ++ct)
            orow[ct * 16 + col] = acc[ct][r];
    }
}

extern "C" void kernel_launch(void* const* d_in, const int* in_sizes, int n_in,
                              void* d_out, int out_size, void* d_ws, size_t ws_size,
                              hipStream_t stream) {
    const float* q    = (const float*)d_in[0];   // [B,N,C]
    const float* k    = (const float*)d_in[1];   // [B,H,N,D]
    const float* v    = (const float*)d_in[2];   // [B,H,N,D]
    const float* w    = (const float*)d_in[3];   // [C,C]
    const float* bias = (const float*)d_in[4];   // [C]
    float* out        = (float*)d_out;           // [B,N,C]
    _Float16* xbuf    = (_Float16*)d_ws;         // 8 MB f16 attention output

    attn_kernel<<<dim3(BB * HH * (NN / QROWS)), dim3(128), 0, stream>>>(q, k, v, xbuf);
    proj_kernel<<<dim3((BB * NN) / 64, CC / 64), dim3(128), 0, stream>>>(xbuf, w, bias, out);
}
